// AttentionPooling_50714973831821
// MI455X (gfx1250) — matrix operand order repaired and checked by reading, not compiled, not run on hardware
//
#include <hip/hip_runtime.h>
#include <hip/hip_bf16.h>
#include <stdint.h>

// ---- problem constants (from reference) ----
#define NNODES  262144
#define DIM     256
#define NHEADS  4
#define HD      64
#define NSEG    4096
#define SCALEF  0.125f      // HEAD_DIM^-0.5 = 1/8
#define EPSF    1e-8f

#define TILE_M  64          // node rows per tile
#define NTILES  (NNODES / TILE_M)
#define RS      264         // padded LDS A row stride (bf16 elems): 528B -> 4-bank step
#define PF      16          // float4 prefetch regs per thread (64 VGPRs)

typedef __attribute__((ext_vector_type(16))) __bf16 v16bf;
typedef __attribute__((ext_vector_type(8)))  __bf16 v8bf;
typedef __attribute__((ext_vector_type(4)))  __bf16 v4bf;
typedef __attribute__((ext_vector_type(8)))  float  v8f;

union AFrag { v16bf v; v8bf h[2]; };
union BFrag { v16bf v; v8bf h[2]; };
union Acc   { v8f   v; float f[8]; };

// ---------------- kernel 0: zero output + attn_sum ----------------
__global__ void k_zero(float* __restrict__ out, float* __restrict__ attn_sum) {
  int i = blockIdx.x * 256 + threadIdx.x;
  if (i < NSEG * DIM) out[i] = 0.0f;
  if (i < NSEG * NHEADS) attn_sum[i] = 0.0f;
}

// ---------------- kernel 1: fold query into key projection ----------------
// w2[h][d] = sum_j query[h][j] * key_w[h*64+j][d];  c[h] = q[h]·key_b[h*64:]
__global__ void k_fold(const float* __restrict__ query, const float* __restrict__ key_w,
                       const float* __restrict__ key_b, float* __restrict__ w2,
                       float* __restrict__ cvec) {
  int d = threadIdx.x;  // 0..255
  for (int h = 0; h < NHEADS; ++h) {
    float s = 0.0f;
    for (int j = 0; j < HD; ++j) s += query[h * HD + j] * key_w[(h * HD + j) * DIM + d];
    w2[h * DIM + d] = s;
  }
  if (d < NHEADS) {
    float s = 0.0f;
    for (int j = 0; j < HD; ++j) s += query[d * HD + j] * key_b[d * HD + j];
    cvec[d] = s;
  }
}

// ---------------- kernel 2: swizzle value_w^T into bf16 B-frag layout ----------------
// Dense bf16 B (32x16, KxN), wave32: lane l holds N = l&15; its 16 contiguous bf16
// elements e are K = 8*(l>>4) + (e&7) + 16*(e>>3) (matches A's table order).
__global__ void k_swz(const float* __restrict__ value_w, __bf16* __restrict__ bsw) {
  int idx = blockIdx.x * 256 + threadIdx.x;     // 65536 elements total
  int e      = idx & 15;
  int lane   = (idx >> 4) & 31;
  int colsub = (idx >> 9) & 15;
  int kk     = idx >> 13;                       // 0..7
  int n = colsub * 16 + (lane & 15);            // output column (B's N)
  int k = kk * 32 + 8 * (lane >> 4) + (e & 7) + 16 * (e >> 3);
  // B = value_w^T : B[k][n] = value_w[n][k]
  bsw[idx] = (__bf16)value_w[n * DIM + k];
}

// ---------------- kernel 3: attention logits -> exp -> segment sums ----------------
__global__ void __launch_bounds__(256) k_attn(
    const float* __restrict__ x, const long long* __restrict__ batch,
    const float* __restrict__ w2, const float* __restrict__ cvec,
    float* __restrict__ attn_exp, float* __restrict__ attn_sum) {
  int lane = threadIdx.x & 31;
  int gw   = (blockIdx.x * blockDim.x + threadIdx.x) >> 5;   // global wave id
  int nw   = (gridDim.x * blockDim.x) >> 5;

  const float4* x4 = (const float4*)x;
  const float4* w4 = (const float4*)w2;

  // loop-invariant per-lane slice of w2 -> registers
  float4 wa[NHEADS], wb[NHEADS];
  float  cs[NHEADS];
#pragma unroll
  for (int h = 0; h < NHEADS; ++h) {
    wa[h] = w4[h * 64 + lane];
    wb[h] = w4[h * 64 + 32 + lane];
    cs[h] = cvec[h];
  }

  for (int node = gw; node < NNODES; node += nw) {
    float4 xa = x4[node * 64 + lane];
    float4 xb = x4[node * 64 + 32 + lane];
    float p[NHEADS];
#pragma unroll
    for (int h = 0; h < NHEADS; ++h) {
      p[h] = xa.x * wa[h].x + xa.y * wa[h].y + xa.z * wa[h].z + xa.w * wa[h].w
           + xb.x * wb[h].x + xb.y * wb[h].y + xb.z * wb[h].z + xb.w * wb[h].w;
    }
#pragma unroll
    for (int off = 16; off >= 1; off >>= 1)
#pragma unroll
      for (int h = 0; h < NHEADS; ++h) p[h] += __shfl_xor(p[h], off);

    if (lane == 0) {
      int sg = (int)batch[node];
#pragma unroll
      for (int h = 0; h < NHEADS; ++h) {
        float e = __expf(SCALEF * (p[h] + cs[h]));
        attn_exp[node * NHEADS + h] = e;
        unsafeAtomicAdd(&attn_sum[sg * NHEADS + h], e);
      }
    }
  }
}

// ---------------- kernel 4: value GEMM (bf16 WMMA) + normalize + segment scatter ----
// LDS: whole bf16 B matrix (128KB, loaded once per persistent block) + 64x256 bf16
// A tile (padded) + per-row segment/coef tables.  ~162KB < 320KB/WGP.
// Software pipelining: next tile's x rows prefetched into 64 VGPRs/thread while the
// current tile's 512 WMMAs execute; B frags loaded one step ahead of each WMMA.
__global__ void __launch_bounds__(256) k_main(
    const float* __restrict__ x, const long long* __restrict__ batch,
    const __bf16* __restrict__ bsw, const float* __restrict__ value_b,
    const float* __restrict__ attn_exp, const float* __restrict__ attn_sum,
    float* __restrict__ out) {
  extern __shared__ char smem[];
  __bf16* ldsB  = (__bf16*)smem;                                 // 131072 B
  __bf16* ldsA  = (__bf16*)(smem + 131072);                      // 64*264*2 = 33792 B
  int*    sSeg  = (int*)(smem + 131072 + 33792);                 // 256 B
  float*  sCoef = (float*)(smem + 131072 + 33792 + 256);         // 1024 B

  int tid   = threadIdx.x;
  int wave  = tid >> 5, lane = tid & 31;
  int lane15 = lane & 15, khalf = lane >> 4;
  int rowsub = wave & 3;          // which 16-row subtile (0..3)
  int ch     = wave >> 2;         // which 128-col half (0..1)

  // stage full B matrix into LDS once (coalesced 16B copies)
  {
    const uint4* src = (const uint4*)bsw;
    uint4*       dst = (uint4*)ldsB;
    for (int i = tid; i < 131072 / 16; i += 256) dst[i] = src[i];
  }

  const float4* x4 = (const float4*)x;

  // prime the pipeline: prefetch first tile's rows into registers
  float4 pre[PF];
  {
    size_t b4 = (size_t)blockIdx.x * TILE_M * 64;   // float4 index of tile base
#pragma unroll
    for (int k = 0; k < PF; ++k) pre[k] = x4[b4 + tid + k * 256];
  }
  __syncthreads();   // covers B staging

  for (int tile = blockIdx.x; tile < NTILES; tile += gridDim.x) {
    int base = tile * TILE_M;

    // ---- drain prefetch regs: fp32 -> bf16 into padded LDS rows ----
#pragma unroll
    for (int k = 0; k < PF; ++k) {
      int i = tid + k * 256;
      int row = i >> 6, c4 = i & 63;
      float4 f = pre[k];
      v4bf hv = { (__bf16)f.x, (__bf16)f.y, (__bf16)f.z, (__bf16)f.w };
      *(v4bf*)(ldsA + row * RS + c4 * 4) = hv;
    }
    // ---- per-row segment id + softmax coefficient ----
    if (tid < TILE_M) {
      int node = base + tid;
      int sg = (int)batch[node];
      sSeg[tid] = sg;
#pragma unroll
      for (int h = 0; h < NHEADS; ++h)
        sCoef[tid * NHEADS + h] =
            attn_exp[node * NHEADS + h] / (attn_sum[sg * NHEADS + h] + EPSF);
    }
    __syncthreads();

    // ---- issue next tile's global loads now; WMMA phase hides their latency ----
    int ntile = tile + gridDim.x;
    if (ntile < NTILES) {
      size_t b4 = (size_t)ntile * TILE_M * 64;
#pragma unroll
      for (int k = 0; k < PF; ++k) pre[k] = x4[b4 + tid + k * 256];
    }

    // ---- WMMA: 16 rows x 128 cols per wave, K=256 in 8 steps of 32 ----
    Acc acc[8];
    {
      v8f z = {0.f, 0.f, 0.f, 0.f, 0.f, 0.f, 0.f, 0.f};
#pragma unroll
      for (int j = 0; j < 8; ++j) acc[j].v = z;
    }

    const __bf16* arow = ldsA + (rowsub * 16 + lane15) * RS + khalf * 8;
    const __bf16* bbase = ldsB + (ch * 8 + (size_t)0) * 32 * 16;  // frag stride 512 elems

    AFrag a;
    a.h[0] = *(const v8bf*)(arow);
    a.h[1] = *(const v8bf*)(arow + 16);

#pragma unroll
    for (int kk = 0; kk < 8; ++kk) {
      AFrag an;
      if (kk < 7) {   // prefetch next A frag
        const __bf16* ap = arow + (kk + 1) * 32;
        an.h[0] = *(const v8bf*)(ap);
        an.h[1] = *(const v8bf*)(ap + 16);
      }
      // B frags pipelined one ahead of each WMMA
      BFrag b;
      b.v = *(const v16bf*)(ldsB + ((kk * 16 + ch * 8 + 0) * 32 + lane) * 16);
#pragma unroll
      for (int j = 0; j < 8; ++j) {
        BFrag bn;
        if (j < 7)
          bn.v = *(const v16bf*)(ldsB + ((kk * 16 + ch * 8 + j + 1) * 32 + lane) * 16);
        acc[j].v = __builtin_amdgcn_wmma_f32_16x16x32_bf16(
            false, a.v, false, b.v, (short)0, acc[j].v, false, false);
        b = bn;
      }
      a = an;
    }
    (void)bbase;

    // ---- epilogue: D layout => element e is row khalf*8+e, col lane15 ----
    int segTop = sSeg[rowsub * 16];
    bool uni = (segTop == sSeg[rowsub * 16 + 15]);   // batch sorted => all equal
#pragma unroll
    for (int j = 0; j < 8; ++j) {
      int col = ch * 128 + j * 16 + lane15;
      int h   = col >> 6;
      float vb = value_b[col];
      if (uni) {
        float s = 0.0f;
#pragma unroll
        for (int e = 0; e < 8; ++e) {
          int r = rowsub * 16 + khalf * 8 + e;
          s += (acc[j].f[e] + vb) * sCoef[r * NHEADS + h];
        }
        s += __shfl_down(s, 16);                      // fold both row-halves
        if (lane < 16) unsafeAtomicAdd(out + segTop * DIM + col, s);
      } else {
#pragma unroll
        for (int e = 0; e < 8; ++e) {
          int r = rowsub * 16 + khalf * 8 + e;
          unsafeAtomicAdd(out + sSeg[r] * DIM + col,
                          (acc[j].f[e] + vb) * sCoef[r * NHEADS + h]);
        }
      }
    }
    __syncthreads();   // LDS reused next tile
  }
}

// ---------------- host side ----------------
extern "C" void kernel_launch(void* const* d_in, const int* in_sizes, int n_in,
                              void* d_out, int out_size, void* d_ws, size_t ws_size,
                              hipStream_t stream) {
  const float*     x       = (const float*)d_in[0];
  const long long* batch   = (const long long*)d_in[1];
  const float*     query   = (const float*)d_in[2];
  const float*     key_w   = (const float*)d_in[3];
  const float*     key_b   = (const float*)d_in[4];
  const float*     value_w = (const float*)d_in[5];
  const float*     value_b = (const float*)d_in[6];
  float*           out     = (float*)d_out;

  char* ws = (char*)d_ws;
  float*  attn_sum = (float*)ws;                                  // 65536 B
  float*  attn_exp = (float*)(ws + 65536);                        // 4 MB
  float*  w2       = (float*)(ws + 65536 + 4194304);              // 4096 B
  float*  cvec     = (float*)(ws + 65536 + 4194304 + 4096);       // 128 B
  __bf16* bsw      = (__bf16*)(ws + 65536 + 4194304 + 4096 + 128);// 128 KB

  k_zero<<<(NSEG * DIM) / 256, 256, 0, stream>>>(out, attn_sum);
  k_fold<<<1, 256, 0, stream>>>(query, key_w, key_b, w2, cvec);
  k_swz<<<(8 * 16 * 32 * 16) / 256, 256, 0, stream>>>(value_w, bsw);
  k_attn<<<2048, 256, 0, stream>>>(x, batch, w2, cvec, attn_exp, attn_sum);

  size_t lds = 131072 + 33792 + 256 + 1024;   // 166144 B
  k_main<<<512, 256, lds, stream>>>(x, batch, bsw, value_b, attn_exp, attn_sum, out);
}
